// RNN2D_70763881168997
// MI455X (gfx1250) — matrix operand order repaired and checked by reading, not compiled, not run on hardware
//
#include <hip/hip_runtime.h>

#define Lr 32
#define Hn 128
#define Bn 512
#define BT 16            // batch rows per workgroup
#define NWG (Bn / BT)    // 32 workgroups
#define EPSv 1e-8f

typedef __attribute__((ext_vector_type(16))) __bf16 v16bf;
typedef __attribute__((ext_vector_type(8)))  __bf16 v8bf;
typedef __attribute__((ext_vector_type(8)))  float  v8f;

// Workspace layout:
//   [0, 32MB)        : Wbf  bf16 [1024][128][128]  = W1[cell][o][k+2] converted
//   [32MB, 32MB+8MB) : hv_g f32  [32 cols][512][128] vertical hidden state
#define WBF_ELEMS ((size_t)Lr * Lr * Hn * Hn)

__device__ __forceinline__ float fast_tanh(float x) {
#if __has_builtin(__builtin_amdgcn_tanhf)
  return __builtin_amdgcn_tanhf(x);      // v_tanh_f32 (gfx1250 TRANS op)
#else
  return tanhf(x);
#endif
}

__device__ __forceinline__ float fast_rcp(float x) {
#if __has_builtin(__builtin_amdgcn_rcpf)
  return __builtin_amdgcn_rcpf(x);       // v_rcp_f32
#else
  return 1.f / x;
#endif
}

// ---------------------------------------------------------------------------
// Pre-pass: convert the K=128 slice of W1 (fp32, [cell][o][130]) to bf16
// [cell][o][128], row-major in K so WMMA B-fragments load contiguously.
// ---------------------------------------------------------------------------
__global__ __launch_bounds__(256) void wconvert_kernel(
    const float* __restrict__ W1, __bf16* __restrict__ Wbf) {
  size_t idx = (size_t)blockIdx.x * blockDim.x + threadIdx.x;
  if (idx >= WBF_ELEMS) return;
  int    k    = (int)(idx & 127);
  size_t oc   = idx >> 7;
  int    o    = (int)(oc & 127);
  size_t cell = oc >> 7;
  Wbf[idx] = (__bf16)W1[(cell * Hn + o) * (Hn + 2) + (k + 2)];
}

// ---------------------------------------------------------------------------
// Main persistent kernel: each WG owns 16 batch rows, walks all 1024 cells
// in zigzag order. Per cell: build inp = h_prev + h_v in bf16 LDS, do
// 16x128 @ 128x128 GEMM with v_wmma_f32_16x16x32_bf16, v_tanh epilogue with
// rank-2 spin correction, parallel fc head + log-prob, store h_v.
// ---------------------------------------------------------------------------
__global__ __launch_bounds__(256) void rnn2d_kernel(
    const float* __restrict__ samples,  // [B][32][32] spins in {0,1}
    const float* __restrict__ W1,       // [1024][128][130] fp32 (cols 0,1 used)
    const float* __restrict__ b1,       // [1024][128]
    const float* __restrict__ Wf,       // [1024][128]
    const float* __restrict__ bfp,      // [1024]
    const __bf16* __restrict__ Wbf,     // [1024][128][128] bf16
    float* __restrict__ hv_g,           // [32][512][128] workspace
    float* __restrict__ out)            // [512]
{
  __shared__ __bf16 s_inp[BT * Hn];     // 4 KB : A-matrix (bf16) for WMMA
  __shared__ float  s_h[BT * Hn];       // 8 KB : current hidden state (f32)
  __shared__ float  s_xs0[BT];          // xh + xv per batch row
  __shared__ float  s_mask[BT];         // current spin per batch row
  __shared__ float  s_wf[Hn];           // fc weights for this cell
  __shared__ float  s_part[BT * 8];     // fc dot partial sums

  const int tid   = threadIdx.x;
  const int bbase = blockIdx.x * BT;
  const int lane  = tid & 31;
  const int wv    = tid >> 5;           // wave id == N tile (0..7)
  const int arow  = lane & 15;          // A-matrix M row for this lane
  const int khalf = lane >> 4;          // which K half this lane holds
  const int ocol  = wv * 16 + (lane & 15); // output column (C/D N index)

  float lp_acc = 0.f;

  for (int r = 0; r < Lr; ++r) {
    const bool odd = (r & 1) != 0;

    // h_h starts at zero each row
    for (int e = tid; e < BT * Hn; e += 256) s_h[e] = 0.f;
    __syncthreads();

    for (int j = 0; j < Lr; ++j) {
      const int c     = odd ? (Lr - 1 - j) : j;       // actual column
      const int cprev = odd ? (c + 1) : (c - 1);      // previously visited col
      const int cell  = r * Lr + c;

      // -------- stage A: inp = bf16(h_prev + h_v), scalars, fc weights ----
      {
        const int e0 = tid * 8;                       // 256 thr * 8 = 2048
        if (r > 0) {
          const float* hvp = hv_g + ((size_t)c * Bn + bbase) * Hn;
          #pragma unroll
          for (int q = 0; q < 8; ++q)
            s_inp[e0 + q] = (__bf16)(s_h[e0 + q] + hvp[e0 + q]);
        } else {
          #pragma unroll
          for (int q = 0; q < 8; ++q)
            s_inp[e0 + q] = (__bf16)s_h[e0 + q];
        }
        if (tid < BT) {
          const float* sb = samples + (size_t)(bbase + tid) * Lr * Lr;
          float xh = (j > 0) ? sb[r * Lr + cprev] : 0.f;
          float xv = (r > 0) ? sb[(r - 1) * Lr + c] : 0.f;
          s_xs0[tid]  = xh + xv;                      // xs1 == 2 - xs0 always
          s_mask[tid] = sb[r * Lr + c];
        }
        if (tid < Hn) s_wf[tid] = Wf[cell * Hn + tid];
      }
      __syncthreads();

      // -------- stage B: WMMA GEMM (16x128x128) + tanh epilogue -----------
      {
        v8f acc = {};
        const __bf16* wrow = Wbf + ((size_t)cell * Hn + ocol) * Hn;
        #pragma unroll
        for (int kb = 0; kb < 4; ++kb) {
          // A fragment: lane holds 16 contiguous K = kb*32+khalf*16..+15
          // for row M = lane%16.
          v16bf afrag = *(const v16bf*)&s_inp[arow * Hn + kb * 32 + khalf * 16];
          // B fragment: lane holds 16 contiguous K of row o = ntile*16+lane%16
          v16bf bfrag = *(const v16bf*)&wrow[kb * 32 + khalf * 16];
          acc = __builtin_amdgcn_wmma_f32_16x16x32_bf16(
              false, afrag, false, bfrag, (short)0, acc, false, false);
        }

        // Prefetch next cell's weight rows and next column's vertical state.
        if (j + 1 < Lr) {
          const int cn = odd ? (Lr - 2 - j) : (j + 1);
          __builtin_prefetch(
              (const void*)(Wbf + ((size_t)(r * Lr + cn) * Hn + ocol) * Hn),
              0, 1);
          if (r > 0)
            __builtin_prefetch(
                (const void*)(hv_g + ((size_t)cn * Bn + bbase) * Hn + tid * 8),
                0, 1);
        }

        // Epilogue: rank-2 spin correction + 2*b1 bias + v_tanh, write s_h.
        const float* w1c = W1 + ((size_t)cell * Hn + ocol) * (Hn + 2);
        const float  w0 = w1c[0], w1e = w1c[1];
        const float  bias2 = 2.f * b1[cell * Hn + ocol];
        #pragma unroll
        for (int i = 0; i < 8; ++i) {
          const int mr = i + (khalf << 3);            // C/D: M = i (+8 hi half)
          const float xs0 = s_xs0[mr];
          const float v = acc[i] + bias2 + xs0 * w0 + (2.f - xs0) * w1e;
          s_h[mr * Hn + ocol] = fast_tanh(v);
        }
      }
      __syncthreads();

      // -------- stage C: parallel fc head + log-prob, store h_v -----------
      if (tid < 128) {                                // 8 partials per batch
        const int b = tid >> 3, q = tid & 7;
        float p = 0.f;
        #pragma unroll
        for (int o = 0; o < 16; ++o)
          p += s_h[b * Hn + q * 16 + o] * s_wf[q * 16 + o];
        s_part[b * 8 + q] = p;
      }
      if (r < Lr - 1) {                               // h_v for next row
        float* hvp = hv_g + ((size_t)c * Bn + bbase) * Hn;
        const int e0 = tid * 8;
        #pragma unroll
        for (int q = 0; q < 8; ++q) hvp[e0 + q] = s_h[e0 + q];
      }
      __syncthreads();

      if (tid < BT) {
        float dot = bfp[cell];
        #pragma unroll
        for (int q = 0; q < 8; ++q) dot += s_part[tid * 8 + q];
        const float xhat = fast_rcp(1.f + __expf(-dot));
        const float m = s_mask[tid];
        lp_acc += __logf(xhat + EPSv) * m + __logf(1.f - xhat + EPSv) * (1.f - m);
      }
      __syncthreads();
    }
  }

  if (tid < BT) out[bbase + tid] = lp_acc;
}

// ---------------------------------------------------------------------------
extern "C" void kernel_launch(void* const* d_in, const int* in_sizes, int n_in,
                              void* d_out, int out_size, void* d_ws, size_t ws_size,
                              hipStream_t stream) {
  const float* samples = (const float*)d_in[0];
  const float* W1      = (const float*)d_in[1];
  const float* b1      = (const float*)d_in[2];
  const float* Wf      = (const float*)d_in[3];
  const float* bfp     = (const float*)d_in[4];

  __bf16* Wbf  = (__bf16*)d_ws;                                      // 32 MB
  float*  hv_g = (float*)((char*)d_ws + WBF_ELEMS * sizeof(__bf16)); // 8 MB
  float*  outp = (float*)d_out;

  const int cvt_blocks = (int)((WBF_ELEMS + 255) / 256);             // 65536
  wconvert_kernel<<<cvt_blocks, 256, 0, stream>>>(W1, Wbf);
  rnn2d_kernel<<<NWG, 256, 0, stream>>>(samples, W1, b1, Wf, bfp,
                                        Wbf, hv_g, outp);
}
